// MPNNLayer_91122026152395
// MI455X (gfx1250) — matrix operand order repaired and checked by reading, not compiled, not run on hardware
//
#include <hip/hip_runtime.h>
#include <cstdint>
#include <cstddef>

typedef __bf16 bf16_t;
typedef __attribute__((ext_vector_type(16))) __bf16 v16bf;
typedef __attribute__((ext_vector_type(8)))  float  v8f;

#define N_NODES  50000
#define DEG      16
#define E_EDGES  (N_NODES * DEG)
#define DIM      128
#define EDGE_DIM 32
#define EPS_LN   1e-5f

// ---- edge-kernel tiling: 1 workgroup = 8 receiver nodes = 128 edges ----
#define NPW   8
#define EPW   (NPW * DEG)            // 128 edges
#define XCOLS (2 * DIM + EDGE_DIM)   // 288
#define XSTR  296                    // padded (592B row = 148 dwords -> conflict-free)
#define HSTR  264                    // padded (528B row = 132 dwords -> conflict-free)

#define XS_BYTES (EPW * XSTR * 2)    // 75776
#define HS_BYTES (EPW * HSTR * 2)    // 67584
#define AS_BYTES (NPW * DIM * 4)     // 4096
#define SMEM_EDGE (XS_BYTES + HS_BYTES + AS_BYTES)  // 147456 B < 320KB WGP LDS

#define NPW2  16                     // nodes per workgroup in node kernel

union Frag {
  v16bf v;
  uint4 q[2];
};

__device__ __forceinline__ v8f wmma_bf16(v16bf a, v16bf b, v8f c) {
  // (neg_a, A, neg_b, B, c_mod, C, reuse_a, reuse_b)
  return __builtin_amdgcn_wmma_f32_16x16x32_bf16(false, a, false, b, (short)0, c,
                                                 false, false);
}

__device__ __forceinline__ float swishf(float x) {
  return x * (1.0f / (1.0f + __expf(-x)));
}

// -------------------- kernel 0: fp32 -> bf16 conversion --------------------
__global__ void cvt_bf16_kernel(const float* __restrict__ src,
                                bf16_t* __restrict__ dst, int n) {
  int i = blockIdx.x * blockDim.x + threadIdx.x;
  if (i < n) dst[i] = (bf16_t)src[i];
}

// ---------- kernel 1: edge MLP + segment-mean + masked LN(agg) ----------
__global__ __launch_bounds__(256) void edge_mlp_kernel(
    const bf16_t* __restrict__ nodeb,
    const float*  __restrict__ edge_feat,
    const int*    __restrict__ senders,
    const bf16_t* __restrict__ We1b, const float* __restrict__ be1,
    const bf16_t* __restrict__ We2b, const float* __restrict__ be2,
    const float*  __restrict__ g_msg, const float* __restrict__ b_msg,
    const float*  __restrict__ node_mask,
    bf16_t* __restrict__ aggb)
{
  extern __shared__ char smem[];
  bf16_t* Xs = (bf16_t*)smem;                       // [128][296] bf16
  bf16_t* Hs = (bf16_t*)(smem + XS_BYTES);          // [128][264] bf16
  float*  As = (float*) (smem + XS_BYTES + HS_BYTES); // [8][128] f32

  const int t    = threadIdx.x;
  const int lane = t & 31;
  const int wave = t >> 5;
  const int wg   = blockIdx.x;
  const int lr   = lane & 15;
  const int kb   = lane >> 4;

  // ---- gather + convert: X tile [128 edges x 288] bf16, 2 threads / row ----
  {
    const int r    = t >> 1;
    const int half = t & 1;
    const size_t e = (size_t)wg * EPW + r;
    if (half == 0) {
      const int snd = senders[e];
      const uint4* s4 = (const uint4*)(nodeb + (size_t)snd * DIM);
      uint4* d4 = (uint4*)(Xs + (size_t)r * XSTR);
      #pragma unroll
      for (int i = 0; i < 16; ++i) d4[i] = s4[i];
    } else {
      const int rcv = wg * NPW + (r >> 4);   // receivers = repeat(arange(N), 16)
      const uint4* s4 = (const uint4*)(nodeb + (size_t)rcv * DIM);
      uint4* d4 = (uint4*)(Xs + (size_t)r * XSTR + DIM);
      #pragma unroll
      for (int i = 0; i < 16; ++i) d4[i] = s4[i];
      const float4* ef = (const float4*)(edge_feat + e * EDGE_DIM);
      bf16_t* xd = Xs + (size_t)r * XSTR + 2 * DIM;
      #pragma unroll
      for (int i = 0; i < 8; ++i) {
        float4 f = ef[i];
        xd[4*i+0] = (bf16_t)f.x; xd[4*i+1] = (bf16_t)f.y;
        xd[4*i+2] = (bf16_t)f.z; xd[4*i+3] = (bf16_t)f.w;
      }
    }
  }
  __syncthreads();

  // ---- stage 1: H = swish(X @ We1 + be1); wave owns N-tiles {2w, 2w+1} ----
  {
    const v8f zero = {};
    v8f C[NPW][2];
    #pragma unroll
    for (int m = 0; m < NPW; ++m) { C[m][0] = zero; C[m][1] = zero; }
    const int n0 = 2 * wave * 16;
    for (int k = 0; k < XCOLS / 32; ++k) {
      Frag b0, b1;
      const bf16_t* bp = We1b + (size_t)(k * 32 + kb * 8) * 256 + n0 + lr;
      #pragma unroll
      for (int s = 0; s < 16; ++s) {
        const int kk = (s & 7) + (s >> 3) * 16;
        b0.v[s] = bp[(size_t)kk * 256];
        b1.v[s] = bp[(size_t)kk * 256 + 16];
      }
      #pragma unroll
      for (int m = 0; m < NPW; ++m) {
        Frag a;                                        // A: row lr, K split by kb
        const bf16_t* ap = Xs + (size_t)(m * 16 + lr) * XSTR + k * 32 + kb * 8;
        a.q[0] = *(const uint4*)(ap);
        a.q[1] = *(const uint4*)(ap + 16);
        C[m][0] = wmma_bf16(a.v, b0.v, C[m][0]);
        C[m][1] = wmma_bf16(a.v, b1.v, C[m][1]);
      }
    }
    const float bias0 = be1[n0 + lr];
    const float bias1 = be1[n0 + 16 + lr];
    #pragma unroll
    for (int m = 0; m < NPW; ++m) {
      #pragma unroll
      for (int v = 0; v < 8; ++v) {
        const int row = m * 16 + kb * 8 + v;           // C row = v + 8*kb
        Hs[(size_t)row * HSTR + n0 + lr]      = (bf16_t)swishf(C[m][0][v] + bias0);
        Hs[(size_t)row * HSTR + n0 + 16 + lr] = (bf16_t)swishf(C[m][1][v] + bias1);
      }
    }
  }
  __syncthreads();

  // ---- stage 2: msgs = H @ We2; M-tile == one node -> in-register mean ----
  {
    const v8f zero = {};
    v8f C2[NPW];
    #pragma unroll
    for (int m = 0; m < NPW; ++m) C2[m] = zero;
    const int n0 = wave * 16;                          // wave owns N-tile w
    for (int k = 0; k < 256 / 32; ++k) {
      Frag b;
      const bf16_t* bp = We2b + (size_t)(k * 32 + kb * 8) * DIM + n0 + lr;
      #pragma unroll
      for (int s = 0; s < 16; ++s)
        b.v[s] = bp[(size_t)((s & 7) + (s >> 3) * 16) * DIM];
      #pragma unroll
      for (int m = 0; m < NPW; ++m) {
        Frag a;
        const bf16_t* ap = Hs + (size_t)(m * 16 + lr) * HSTR + k * 32 + kb * 8;
        a.q[0] = *(const uint4*)(ap);
        a.q[1] = *(const uint4*)(ap + 16);
        C2[m] = wmma_bf16(a.v, b.v, C2[m]);
      }
    }
    const float b2 = be2[n0 + lr];
    #pragma unroll
    for (int m = 0; m < NPW; ++m) {
      float s = 0.f;
      #pragma unroll
      for (int v = 0; v < 8; ++v) s += C2[m][v];       // 8 rows in this half
      s += __shfl_xor(s, 16, 32);                      // + other 8 rows
      if (lane < 16) As[m * DIM + n0 + lr] = s * (1.0f / DEG) + b2;
    }
  }
  __syncthreads();

  // ---- masked LayerNorm of agg; one wave per node; write bf16 ----
  {
    const int node  = wave;
    const int gnode = wg * NPW + node;
    const float x0 = As[node * DIM + lane];
    const float x1 = As[node * DIM + lane + 32];
    const float x2 = As[node * DIM + lane + 64];
    const float x3 = As[node * DIM + lane + 96];
    float s = x0 + x1 + x2 + x3;
    #pragma unroll
    for (int o = 16; o > 0; o >>= 1) s += __shfl_xor(s, o, 32);
    const float mean = s * (1.0f / DIM);
    const float d0 = x0 - mean, d1 = x1 - mean, d2 = x2 - mean, d3 = x3 - mean;
    float q = d0*d0 + d1*d1 + d2*d2 + d3*d3;
    #pragma unroll
    for (int o = 16; o > 0; o >>= 1) q += __shfl_xor(q, o, 32);
    const float rinv = rsqrtf(q * (1.0f / DIM) + EPS_LN);
    const float mk = node_mask[gnode];
    bf16_t* dst = aggb + (size_t)gnode * DIM;
    dst[lane]      = (bf16_t)((d0 * rinv * g_msg[lane]      + b_msg[lane])      * mk);
    dst[lane + 32] = (bf16_t)((d1 * rinv * g_msg[lane + 32] + b_msg[lane + 32]) * mk);
    dst[lane + 64] = (bf16_t)((d2 * rinv * g_msg[lane + 64] + b_msg[lane + 64]) * mk);
    dst[lane + 96] = (bf16_t)((d3 * rinv * g_msg[lane + 96] + b_msg[lane + 96]) * mk);
  }
}

// ---------- kernel 2: node MLP + residual + masked LN ----------
__global__ __launch_bounds__(256) void node_mlp_kernel(
    const float*  __restrict__ node_inp,
    const bf16_t* __restrict__ nodeb,
    const bf16_t* __restrict__ aggb,
    const bf16_t* __restrict__ Wn1b, const float* __restrict__ bn1,
    const bf16_t* __restrict__ Wn2b, const float* __restrict__ bn2,
    const float*  __restrict__ g_node, const float* __restrict__ b_node,
    const float*  __restrict__ node_mask,
    float* __restrict__ out)
{
  __shared__ bf16_t Ys [NPW2 * HSTR];   // [16][264] bf16
  __shared__ bf16_t H2s[NPW2 * HSTR];   // [16][264] bf16
  __shared__ float  Os [NPW2 * DIM];    // [16][128] f32

  const int t = threadIdx.x;
  const int lane = t & 31;
  const int wave = t >> 5;
  const int wg = blockIdx.x;
  const int nb = wg * NPW2;
  const int lr = lane & 15;
  const int kb = lane >> 4;

  // Y = concat(nodeb, aggb) : 16 rows x 256 cols bf16 (32B per thread)
  {
    const int r = t >> 4;
    const int c = t & 15;
    const bf16_t* src = (c < 8)
        ? (nodeb + (size_t)(nb + r) * DIM + c * 16)
        : (aggb  + (size_t)(nb + r) * DIM + (c - 8) * 16);
    uint4* d4 = (uint4*)(Ys + (size_t)r * HSTR + c * 16);
    const uint4* s4 = (const uint4*)src;
    d4[0] = s4[0]; d4[1] = s4[1];
  }
  __syncthreads();

  // stage 1: H2 = swish(Y @ Wn1 + bn1); wave owns N-tiles {2w, 2w+1}
  {
    v8f C0 = {}, C1 = {};
    const int n0 = 2 * wave * 16;
    for (int k = 0; k < 8; ++k) {
      Frag a, b0, b1;
      const bf16_t* ap = Ys + (size_t)lr * HSTR + k * 32 + kb * 8;
      a.q[0] = *(const uint4*)(ap);
      a.q[1] = *(const uint4*)(ap + 16);
      const bf16_t* bp = Wn1b + (size_t)(k * 32 + kb * 8) * 256 + n0 + lr;
      #pragma unroll
      for (int s = 0; s < 16; ++s) {
        const int kk = (s & 7) + (s >> 3) * 16;
        b0.v[s] = bp[(size_t)kk * 256];
        b1.v[s] = bp[(size_t)kk * 256 + 16];
      }
      C0 = wmma_bf16(a.v, b0.v, C0);
      C1 = wmma_bf16(a.v, b1.v, C1);
    }
    const float bias0 = bn1[n0 + lr];
    const float bias1 = bn1[n0 + 16 + lr];
    #pragma unroll
    for (int v = 0; v < 8; ++v) {
      const int row = kb * 8 + v;
      H2s[(size_t)row * HSTR + n0 + lr]      = (bf16_t)swishf(C0[v] + bias0);
      H2s[(size_t)row * HSTR + n0 + 16 + lr] = (bf16_t)swishf(C1[v] + bias1);
    }
  }
  __syncthreads();

  // stage 2: nf = node_inp + H2 @ Wn2 + bn2 ; wave owns N-tile w
  {
    v8f C2 = {};
    const int n0 = wave * 16;
    for (int k = 0; k < 8; ++k) {
      Frag a, b;
      const bf16_t* ap = H2s + (size_t)lr * HSTR + k * 32 + kb * 8;
      a.q[0] = *(const uint4*)(ap);
      a.q[1] = *(const uint4*)(ap + 16);
      const bf16_t* bp = Wn2b + (size_t)(k * 32 + kb * 8) * DIM + n0 + lr;
      #pragma unroll
      for (int s = 0; s < 16; ++s)
        b.v[s] = bp[(size_t)((s & 7) + (s >> 3) * 16) * DIM];
      C2 = wmma_bf16(a.v, b.v, C2);
    }
    const float b2 = bn2[n0 + lr];
    #pragma unroll
    for (int v = 0; v < 8; ++v) {
      const int row = kb * 8 + v;
      Os[row * DIM + n0 + lr] =
          C2[v] + b2 + node_inp[(size_t)(nb + row) * DIM + n0 + lr];
    }
  }
  __syncthreads();

  // final masked LN; 8 waves x 2 nodes, fp32 out
  #pragma unroll
  for (int i = 0; i < 2; ++i) {
    const int node = wave * 2 + i;
    const float x0 = Os[node * DIM + lane];
    const float x1 = Os[node * DIM + lane + 32];
    const float x2 = Os[node * DIM + lane + 64];
    const float x3 = Os[node * DIM + lane + 96];
    float s = x0 + x1 + x2 + x3;
    #pragma unroll
    for (int o = 16; o > 0; o >>= 1) s += __shfl_xor(s, o, 32);
    const float mean = s * (1.0f / DIM);
    const float d0 = x0 - mean, d1 = x1 - mean, d2 = x2 - mean, d3 = x3 - mean;
    float q = d0*d0 + d1*d1 + d2*d2 + d3*d3;
    #pragma unroll
    for (int o = 16; o > 0; o >>= 1) q += __shfl_xor(q, o, 32);
    const float rinv = rsqrtf(q * (1.0f / DIM) + EPS_LN);
    const float mk = node_mask[nb + node];
    float* dst = out + (size_t)(nb + node) * DIM;
    dst[lane]      = (d0 * rinv * g_node[lane]      + b_node[lane])      * mk;
    dst[lane + 32] = (d1 * rinv * g_node[lane + 32] + b_node[lane + 32]) * mk;
    dst[lane + 64] = (d2 * rinv * g_node[lane + 64] + b_node[lane + 64]) * mk;
    dst[lane + 96] = (d3 * rinv * g_node[lane + 96] + b_node[lane + 96]) * mk;
  }
}

// ---------------------------- launch ----------------------------
extern "C" void kernel_launch(void* const* d_in, const int* in_sizes, int n_in,
                              void* d_out, int out_size, void* d_ws, size_t ws_size,
                              hipStream_t stream) {
  (void)in_sizes; (void)n_in; (void)out_size; (void)ws_size;
  const float* node_inp  = (const float*)d_in[0];
  const float* edge_feat = (const float*)d_in[1];
  const float* node_mask = (const float*)d_in[2];
  const float* We1  = (const float*)d_in[3];
  const float* be1  = (const float*)d_in[4];
  const float* We2  = (const float*)d_in[5];
  const float* be2  = (const float*)d_in[6];
  const float* g_msg = (const float*)d_in[7];
  const float* b_msg = (const float*)d_in[8];
  const float* Wn1  = (const float*)d_in[9];
  const float* bn1  = (const float*)d_in[10];
  const float* Wn2  = (const float*)d_in[11];
  const float* bn2  = (const float*)d_in[12];
  const float* g_node = (const float*)d_in[13];
  const float* b_node = (const float*)d_in[14];
  const int*   senders = (const int*)d_in[15];
  // d_in[16] (receivers) implicit: repeat(arange(N), DEG)
  float* out = (float*)d_out;

  // workspace layout (bf16 staging), all offsets 256B-aligned
  char* ws = (char*)d_ws;
  bf16_t* nodeb = (bf16_t*)(ws);
  bf16_t* We1b  = (bf16_t*)(ws + 12800000);
  bf16_t* We2b  = (bf16_t*)(ws + 12800000 + 147456);
  bf16_t* Wn1b  = (bf16_t*)(ws + 12800000 + 147456 + 65536);
  bf16_t* Wn2b  = (bf16_t*)(ws + 12800000 + 147456 + 65536 + 131072);
  bf16_t* aggb  = (bf16_t*)(ws + 12800000 + 147456 + 65536 + 131072 + 65536);

  auto cvt = [&](const float* s, bf16_t* d, int n) {
    cvt_bf16_kernel<<<(n + 255) / 256, 256, 0, stream>>>(s, d, n);
  };
  cvt(node_inp, nodeb, N_NODES * DIM);
  cvt(We1, We1b, XCOLS * 256);
  cvt(We2, We2b, 256 * DIM);
  cvt(Wn1, Wn1b, 256 * 256);
  cvt(Wn2, Wn2b, 256 * DIM);

  (void)hipFuncSetAttribute((const void*)edge_mlp_kernel,
                            hipFuncAttributeMaxDynamicSharedMemorySize, SMEM_EDGE);
  edge_mlp_kernel<<<E_EDGES / EPW, 256, SMEM_EDGE, stream>>>(
      nodeb, edge_feat, senders, We1b, be1, We2b, be2, g_msg, b_msg, node_mask,
      aggb);

  node_mlp_kernel<<<N_NODES / NPW2, 256, 0, stream>>>(
      node_inp, nodeb, aggb, Wn1b, bn1, Wn2b, bn2, g_node, b_node, node_mask,
      out);
}
